// NeighborhoodAugmenter_21414706938291
// MI455X (gfx1250) — matrix operand order repaired and checked by reading, not compiled, not run on hardware
//
#include <hip/hip_runtime.h>

// NeighborhoodAugmenter for MI455X (gfx1250, wave32, WMMA).
//
// Pipeline:
//   K1: h_norm (f16) = latent / ||latent||          -> ws[0 .. 1MB)
//   K2: fused  sim = h_norm @ h_norm^T  (v_wmma_f32_16x16x32_f16)
//       + diagonal mask + running top-3 + rand pick -> nb_idx in ws[1MB ..)
//   K3: out = (unif < 0.8) ? x : x[nb]   (float4 streaming, HBM-bound)

#define B_ROWS 8192
#define G_COLS 20000
#define D_DIM  64
#define MIXR   0.8f
#define NEGF   (-9e15f)

typedef _Float16 half8 __attribute__((ext_vector_type(8)));
typedef _Float16 v16h  __attribute__((ext_vector_type(16)));
typedef float    v8f   __attribute__((ext_vector_type(8)));

union V16U { v16h v; half8 h[2]; };

// ---------------------------------------------------------------------------
// Kernel 1: one wave (32 lanes) per row, 2 floats per lane, shfl reduction.
// ---------------------------------------------------------------------------
__global__ void norm_rows_f16(const float* __restrict__ latent,
                              _Float16* __restrict__ hn) {
  const int wave = (blockIdx.x * blockDim.x + threadIdx.x) >> 5;
  const int lane = threadIdx.x & 31;
  if (wave >= B_ROWS) return;
  const float* row = latent + (size_t)wave * D_DIM;
  float a0 = row[lane * 2 + 0];
  float a1 = row[lane * 2 + 1];
  float s = a0 * a0 + a1 * a1;
  s += __shfl_xor(s, 1);
  s += __shfl_xor(s, 2);
  s += __shfl_xor(s, 4);
  s += __shfl_xor(s, 8);
  s += __shfl_xor(s, 16);
  const float inv = rsqrtf(s);
  union { _Float16 h[2]; unsigned u; } pk;
  pk.h[0] = (_Float16)(a0 * inv);
  pk.h[1] = (_Float16)(a1 * inv);
  ((unsigned*)(hn + (size_t)wave * D_DIM))[lane] = pk.u;
}

// ---------------------------------------------------------------------------
// Kernel 2: one wave per 16-row strip of sim. WMMA f32_16x16x32_f16, K=64.
//
// A layout (16x32, contiguous k order, same as B): lane L holds row M = L%16; half = L/16;
//   a[0..15] = h_norm[M][ half*16 + 0..15 ]
// B layout (K x N, per sparse-B analogy): lane L holds col N = L%16;
//   b[0..15] = h_norm[N][ half*16 + 0..15 ]          (B[k,n] = h_norm[n][k])
// C layout: VGPR c, lanes 0-15: (M=c,     N=lane)
//           VGPR c, lanes16-31: (M=c+8,   N=lane-16)
// ---------------------------------------------------------------------------
__global__ void __launch_bounds__(32)
topk_neighbors(const _Float16* __restrict__ hn,
               const int* __restrict__ rand_idx,
               int* __restrict__ nb_out) {
  __shared__ float lds[16][33];  // 16 rows x 32 cols, stride 33: conflict-free

  const int lane     = threadIdx.x & 31;
  const int r        = lane & 15;
  const int half     = lane >> 4;
  const int row_base = blockIdx.x * 16;
  const int grow     = row_base + r;   // global sim-row this lane owns

  // Load A tile once (K 0..31 and K 32..63)
  const _Float16* ap = hn + (size_t)grow * D_DIM;
  V16U alo, ahi;
  alo.h[0] = *(const half8*)(ap +  0 + half * 16);
  alo.h[1] = *(const half8*)(ap +  8 + half * 16);
  ahi.h[0] = *(const half8*)(ap + 32 + half * 16);
  ahi.h[1] = *(const half8*)(ap + 40 + half * 16);

  float v0 = NEGF, v1 = NEGF, v2 = NEGF;
  int   i0 = 0,    i1 = 0,    i2 = 0;

  for (int cb = 0; cb < B_ROWS; cb += 32) {
    const _Float16* bp0 = hn + (size_t)(cb + r)      * D_DIM;
    const _Float16* bp1 = hn + (size_t)(cb + 16 + r) * D_DIM;
    V16U b0lo, b0hi, b1lo, b1hi;
    b0lo.h[0] = *(const half8*)(bp0 +  0 + half * 16);
    b0lo.h[1] = *(const half8*)(bp0 +  8 + half * 16);
    b0hi.h[0] = *(const half8*)(bp0 + 32 + half * 16);
    b0hi.h[1] = *(const half8*)(bp0 + 40 + half * 16);
    b1lo.h[0] = *(const half8*)(bp1 +  0 + half * 16);
    b1lo.h[1] = *(const half8*)(bp1 +  8 + half * 16);
    b1hi.h[0] = *(const half8*)(bp1 + 32 + half * 16);
    b1hi.h[1] = *(const half8*)(bp1 + 40 + half * 16);

    v8f c0 = {}; v8f c1 = {};
    c0 = __builtin_amdgcn_wmma_f32_16x16x32_f16(false, alo.v, false, b0lo.v,
                                                (short)0, c0, false, false);
    c0 = __builtin_amdgcn_wmma_f32_16x16x32_f16(false, ahi.v, false, b0hi.v,
                                                (short)0, c0, false, false);
    c1 = __builtin_amdgcn_wmma_f32_16x16x32_f16(false, alo.v, false, b1lo.v,
                                                (short)0, c1, false, false);
    c1 = __builtin_amdgcn_wmma_f32_16x16x32_f16(false, ahi.v, false, b1hi.v,
                                                (short)0, c1, false, false);

    // Spill C tiles (cols cb+0..15 from c0, cb+16..31 from c1) to LDS.
#pragma unroll
    for (int c = 0; c < 8; ++c) {
      lds[c + 8 * half][r]      = c0[c];
      lds[c + 8 * half][16 + r] = c1[c];
    }
    __syncthreads();  // single-wave WG: s_barrier ~ NOP; forces LDS ordering

    // Each lane scans 16 columns of its row; branch-free top-3 insert so the
    // wave re-enters the WMMAs with EXEC all-ones.
#pragma unroll
    for (int s = 0; s < 16; ++s) {
      const int col = cb + half * 16 + s;
      float v = lds[r][half * 16 + s];
      v = (col == grow) ? NEGF : v;          // diagonal mask
      const bool g0 = v > v0, g1 = v > v1, g2 = v > v2;
      const float nv2 = g1 ? v1 : (g2 ? v : v2);
      const int   ni2 = g1 ? i1 : (g2 ? col : i2);
      const float nv1 = g0 ? v0 : (g1 ? v : v1);
      const int   ni1 = g0 ? i0 : (g1 ? col : i1);
      const float nv0 = g0 ? v  : v0;
      const int   ni0 = g0 ? col : i0;
      v0 = nv0; v1 = nv1; v2 = nv2;
      i0 = ni0; i1 = ni1; i2 = ni2;
    }
    __syncthreads();
  }

  // Merge high-half candidates into low-half top-3 through LDS.
  if (half) {
    lds[r][0] = v0; lds[r][1] = __int_as_float(i0);
    lds[r][2] = v1; lds[r][3] = __int_as_float(i1);
    lds[r][4] = v2; lds[r][5] = __int_as_float(i2);
  }
  __syncthreads();
  if (!half) {
#pragma unroll
    for (int j = 0; j < 3; ++j) {
      const float v   = lds[r][2 * j];
      const int   col = __float_as_int(lds[r][2 * j + 1]);
      const bool g0 = v > v0, g1 = v > v1, g2 = v > v2;
      const float nv2 = g1 ? v1 : (g2 ? v : v2);
      const int   ni2 = g1 ? i1 : (g2 ? col : i2);
      const float nv1 = g0 ? v0 : (g1 ? v : v1);
      const int   ni1 = g0 ? i0 : (g1 ? col : i1);
      const float nv0 = g0 ? v  : v0;
      const int   ni0 = g0 ? col : i0;
      v0 = nv0; v1 = nv1; v2 = nv2;
      i0 = ni0; i1 = ni1; i2 = ni2;
    }
    const int k = rand_idx[grow];
    nb_out[grow] = (k == 0) ? i0 : ((k == 1) ? i1 : i2);
  }
}

// ---------------------------------------------------------------------------
// Kernel 3: HBM-bound mix. G = 20000 floats per row = 5000 float4 (16B rows).
// ---------------------------------------------------------------------------
__global__ void mix_swap(const float4* __restrict__ x,
                         const float4* __restrict__ unif,
                         const int* __restrict__ nb_idx,
                         float4* __restrict__ out) {
  const int G4 = G_COLS / 4;
  const int b  = blockIdx.y;
  const int g  = blockIdx.x * blockDim.x + threadIdx.x;
  if (g >= G4) return;
  const int n = nb_idx[b];               // uniform per block -> scalar load
  const size_t self = (size_t)b * G4 + g;
  const float4 xa = x[self];
  const float4 xn = x[(size_t)n * G4 + g];
  const float4 u  = unif[self];
  float4 o;
  o.x = (u.x < MIXR) ? xa.x : xn.x;
  o.y = (u.y < MIXR) ? xa.y : xn.y;
  o.z = (u.z < MIXR) ? xa.z : xn.z;
  o.w = (u.w < MIXR) ? xa.w : xn.w;
  out[self] = o;
}

// ---------------------------------------------------------------------------
extern "C" void kernel_launch(void* const* d_in, const int* in_sizes, int n_in,
                              void* d_out, int out_size, void* d_ws,
                              size_t ws_size, hipStream_t stream) {
  const float* x        = (const float*)d_in[0];  // [8192, 20000] f32
  const float* latent   = (const float*)d_in[1];  // [8192, 64]    f32
  const int*   rand_idx = (const int*)d_in[2];    // [8192]        i32
  const float* unif     = (const float*)d_in[3];  // [8192, 20000] f32
  float*       out      = (float*)d_out;

  _Float16* hn = (_Float16*)d_ws;                          // 1 MB
  int* nb = (int*)((char*)d_ws + (size_t)B_ROWS * D_DIM * sizeof(_Float16));

  // K1: 8192 waves = 1024 blocks x 256 threads
  norm_rows_f16<<<B_ROWS / 8, 256, 0, stream>>>(latent, hn);

  // K2: one wave per 16-row strip -> 512 single-wave workgroups
  topk_neighbors<<<B_ROWS / 16, 32, 0, stream>>>(hn, rand_idx, nb);

  // K3: streaming mix
  dim3 g3((G_COLS / 4 + 255) / 256, B_ROWS);
  mix_swap<<<g3, 256, 0, stream>>>((const float4*)x, (const float4*)unif, nb,
                                   (float4*)out);
}
